// LlamaAttention_67190468379096
// MI455X (gfx1250) — compile-verified, matrix operand order checked by reading, not run on hardware
//
#include <hip/hip_runtime.h>

#define S_LEN 4096
#define HID   2048
#define NH    16
#define NKV   4
#define HD    128
#define GRP   (NH / NKV)

typedef __attribute__((ext_vector_type(8)))  __bf16 v8bf;
typedef __attribute__((ext_vector_type(16))) __bf16 v16bf;
typedef __attribute__((ext_vector_type(8)))  float  v8f;

// Load one 16x32 bf16 WMMA fragment for this lane. Caller passes the lane's
// base pointer: row_base + (lane&15) rows in, plus (lane>=16 ? 8 : 0) halves.
// Layout per CDNA5 ISA 7.12.2: lane<16 holds K {0..7,16..23}, lane>=16 holds
// K {8..15,24..31} of its row — i.e. two contiguous 16B chunks 32B apart.
static __device__ __forceinline__ v16bf load_frag(const __bf16* p) {
    v8bf lo = *(const v8bf*)p;
    v8bf hi = *(const v8bf*)(p + 16);
    return __builtin_shufflevector(lo, hi, 0, 1, 2, 3, 4, 5, 6, 7,
                                           8, 9, 10, 11, 12, 13, 14, 15);
}

static __device__ __forceinline__ v8f wmma_bf16(v16bf a, v16bf b, v8f c) {
    return __builtin_amdgcn_wmma_f32_16x16x32_bf16(false, a, false, b,
                                                   (short)0, c, false, false);
}

// ---------------------------------------------------------------- conversions
__global__ void f32_to_bf16_kernel(const float* __restrict__ in,
                                   __bf16* __restrict__ out, long n) {
    long i = (long)blockIdx.x * blockDim.x + threadIdx.x;
    long stride = (long)gridDim.x * blockDim.x;
    for (; i < n; i += stride) out[i] = (__bf16)in[i];
}

// ------------------------------------------------------------------- GEMM
// C[M,N] = A[M,K] * W[N,K]^T  (A,W bf16 row-major; C OutT row-major)
// Block = 128x128 tile, 8 waves (4 along M x 2 along N), wave = 32x64 = 8 WMMA
// accumulators. M,N assumed multiples of 128, K multiple of 32 (true here),
// so EXEC stays all-ones around every WMMA.
template <typename OutT>
__global__ __launch_bounds__(256)
void gemm_xwT(const __bf16* __restrict__ A, const __bf16* __restrict__ W,
              OutT* __restrict__ C, int N, int K) {
    const int wave = threadIdx.x >> 5;
    const int lane = threadIdx.x & 31;
    const int half = lane >> 4;
    const int r    = lane & 15;
    const int wm = blockIdx.y * 128 + (wave >> 1) * 32;
    const int wn = blockIdx.x * 128 + (wave & 1) * 64;

    const v8f zero = {0, 0, 0, 0, 0, 0, 0, 0};
    v8f acc[2][4];
#pragma unroll
    for (int i = 0; i < 2; i++)
#pragma unroll
        for (int j = 0; j < 4; j++) acc[i][j] = zero;

    for (int k0 = 0; k0 < K; k0 += 32) {
        v16bf a[2], b[4];
#pragma unroll
        for (int i = 0; i < 2; i++)
            a[i] = load_frag(A + (size_t)(wm + i * 16 + r) * K + k0 + half * 8);
#pragma unroll
        for (int j = 0; j < 4; j++)
            b[j] = load_frag(W + (size_t)(wn + j * 16 + r) * K + k0 + half * 8);
#pragma unroll
        for (int i = 0; i < 2; i++)
#pragma unroll
            for (int j = 0; j < 4; j++)
                acc[i][j] = wmma_bf16(a[i], b[j], acc[i][j]);
    }
    // C fragment layout: VGPR e -> row e + 8*half, col = lane&15.
#pragma unroll
    for (int i = 0; i < 2; i++)
#pragma unroll
        for (int j = 0; j < 4; j++)
#pragma unroll
            for (int e = 0; e < 8; e++) {
                int row = wm + i * 16 + e + half * 8;
                int col = wn + j * 16 + r;
                C[(size_t)row * N + col] = (OutT)acc[i][j][e];
            }
}

// --------------------------------------------------------------------- RoPE
// raw: [S, nheads*HD] bf16 -> out: [nheads, S, HD] bf16 (head-major for attn)
__global__ __launch_bounds__(64)
void rope_kernel(const __bf16* __restrict__ raw, __bf16* __restrict__ out,
                 const int* __restrict__ pos, int nheads) {
    const int s = blockIdx.x;
    const int h = blockIdx.y;
    const int d = threadIdx.x;  // 0..63
    const float p = (float)pos[s];
    // inv_freq[d] = 10000^(-d/64) = exp(-d * ln(10000)/64)
    const float ang = p * __expf(-0.14391156642f * (float)d);
    float sn, cs;
    __sincosf(ang, &sn, &cs);
    const __bf16* rowp = raw + (size_t)s * (nheads * HD) + h * HD;
    const float x1 = (float)rowp[d];
    const float x2 = (float)rowp[d + 64];
    __bf16* orow = out + ((size_t)h * S_LEN + s) * HD;
    orow[d]      = (__bf16)(x1 * cs - x2 * sn);
    orow[d + 64] = (__bf16)(x2 * cs + x1 * sn);
}

// V: [S, NKV*HD] -> Vt: [NKV, HD, S] so P*V B-fragments load contiguous rows.
__global__ void vtrans_kernel(const __bf16* __restrict__ raw,
                              __bf16* __restrict__ out) {
    long idx = (long)blockIdx.x * blockDim.x + threadIdx.x;
    if (idx >= (long)S_LEN * NKV * HD) return;
    int s = (int)(idx / (NKV * HD));
    int rem = (int)(idx % (NKV * HD));
    int h = rem / HD, d = rem % HD;
    out[((size_t)h * HD + d) * S_LEN + s] = raw[idx];
}

// ----------------------------------------------------- flash attention (GQA)
// One wave32 per (head, 16-row query tile). Causal mask applied analytically.
__global__ __launch_bounds__(32)
void attn_kernel(const __bf16* __restrict__ Qb, const __bf16* __restrict__ Kb,
                 const __bf16* __restrict__ Vtb, __bf16* __restrict__ AttnB) {
    __shared__ __bf16 ldsP[16 * 32];  // P tile bounce buffer (C-layout -> A-layout)
    const int h = blockIdx.x;
    const int qBase = blockIdx.y * 16;
    const int hk = h / GRP;
    const int lane = threadIdx.x & 31;
    const int half = lane >> 4;
    const int r = lane & 15;
    const float scale = 0.08838834764831845f;  // 1/sqrt(128)

    const __bf16* Qh = Qb + (size_t)h * S_LEN * HD;
    const __bf16* Kh = Kb + (size_t)hk * S_LEN * HD;
    const __bf16* Vh = Vtb + (size_t)hk * HD * S_LEN;

    v16bf qa[4];
#pragma unroll
    for (int c = 0; c < 4; c++)
        qa[c] = load_frag(Qh + (size_t)(qBase + r) * HD + c * 32 + half * 8);

    const v8f zero = {0, 0, 0, 0, 0, 0, 0, 0};
    v8f o[8];
#pragma unroll
    for (int t = 0; t < 8; t++) o[t] = zero;
    float mrow[8], lrow[8];
#pragma unroll
    for (int i = 0; i < 8; i++) { mrow[i] = -3.0e38f; lrow[i] = 0.0f; }

    // Key tiles of 32; last tile jb<=qBase covers keys up to qBase+31>=qBase+15.
    for (int jb = 0; jb <= qBase; jb += 32) {
        v8f s0 = zero, s1 = zero;
#pragma unroll
        for (int c = 0; c < 4; c++) {
            v16bf kb0 = load_frag(Kh + (size_t)(jb + r) * HD + c * 32 + half * 8);
            v16bf kb1 = load_frag(Kh + (size_t)(jb + 16 + r) * HD + c * 32 + half * 8);
            s0 = wmma_bf16(qa[c], kb0, s0);
            s1 = wmma_bf16(qa[c], kb1, s1);
        }
        // Online softmax. Row m = i + 8*half lives in 16 lanes of one half,
        // so intra-half xor-shuffles (masks 1,2,4,8) reduce across columns.
        float p0[8], p1[8];
#pragma unroll
        for (int i = 0; i < 8; i++) {
            const int qrow = qBase + i + half * 8;
            float v0 = (jb + r)      <= qrow ? s0[i] * scale : -3.0e38f;
            float v1 = (jb + 16 + r) <= qrow ? s1[i] * scale : -3.0e38f;
            float rmax = fmaxf(v0, v1);
            rmax = fmaxf(rmax, __shfl_xor(rmax, 1, 32));
            rmax = fmaxf(rmax, __shfl_xor(rmax, 2, 32));
            rmax = fmaxf(rmax, __shfl_xor(rmax, 4, 32));
            rmax = fmaxf(rmax, __shfl_xor(rmax, 8, 32));
            const float mnew = fmaxf(mrow[i], rmax);
            const float e0 = __expf(v0 - mnew);
            const float e1 = __expf(v1 - mnew);
            float rsum = e0 + e1;
            rsum += __shfl_xor(rsum, 1, 32);
            rsum += __shfl_xor(rsum, 2, 32);
            rsum += __shfl_xor(rsum, 4, 32);
            rsum += __shfl_xor(rsum, 8, 32);
            const float alpha = __expf(mrow[i] - mnew);
            lrow[i] = lrow[i] * alpha + rsum;
            mrow[i] = mnew;
            p0[i] = e0;
            p1[i] = e1;
#pragma unroll
            for (int t = 0; t < 8; t++) o[t][i] *= alpha;
        }
        // C-layout P -> LDS -> A-layout fragment for the P*V WMMA.
#pragma unroll
        for (int i = 0; i < 8; i++) {
            ldsP[(i + half * 8) * 32 + r]      = (__bf16)p0[i];
            ldsP[(i + half * 8) * 32 + 16 + r] = (__bf16)p1[i];
        }
        __syncthreads();
        const v16bf pa = load_frag(ldsP + r * 32 + half * 8);
        __syncthreads();
#pragma unroll
        for (int t = 0; t < 8; t++) {
            v16bf vb = load_frag(Vh + (size_t)(t * 16 + r) * S_LEN + jb + half * 8);
            o[t] = wmma_bf16(pa, vb, o[t]);
        }
    }
    // Normalize and emit attention output in [S, NH*HD] bf16 for the O-proj GEMM.
#pragma unroll
    for (int t = 0; t < 8; t++)
#pragma unroll
        for (int i = 0; i < 8; i++) {
            const int row = qBase + i + half * 8;
            const int col = h * HD + t * 16 + r;
            AttnB[(size_t)row * (NH * HD) + col] = (__bf16)(o[t][i] / lrow[i]);
        }
}

// ------------------------------------------------------------------ launcher
extern "C" void kernel_launch(void* const* d_in, const int* in_sizes, int n_in,
                              void* d_out, int out_size, void* d_ws, size_t ws_size,
                              hipStream_t stream) {
    (void)in_sizes; (void)n_in; (void)out_size;
    const float* hs = (const float*)d_in[0];
    // d_in[1] attention_mask: exactly causal -> applied analytically in attn_kernel
    const int*   pos = (const int*)d_in[2];
    const float* Wq  = (const float*)d_in[3];
    const float* Wk  = (const float*)d_in[4];
    const float* Wv  = (const float*)d_in[5];
    const float* Wo  = (const float*)d_in[6];
    float* out = (float*)d_out;

    const size_t MB = 1024 * 1024;
    if (ws_size < 84 * MB) return;  // need 84 MB scratch
    char* ws = (char*)d_ws;
    __bf16* Xb   = (__bf16*)(ws + 0);        // 16 MB  [S, HID]
    __bf16* Wqb  = (__bf16*)(ws + 16 * MB);  //  8 MB
    __bf16* Wkb  = (__bf16*)(ws + 24 * MB);  //  2 MB
    __bf16* Wvb  = (__bf16*)(ws + 26 * MB);  //  2 MB
    __bf16* Wob  = (__bf16*)(ws + 28 * MB);  //  8 MB
    __bf16* Qraw = (__bf16*)(ws + 36 * MB);  // 16 MB  [S, NH*HD]   (reused as AttnB)
    __bf16* Kraw = (__bf16*)(ws + 52 * MB);  //  4 MB  [S, NKV*HD]
    __bf16* Vraw = (__bf16*)(ws + 56 * MB);  //  4 MB  [S, NKV*HD]
    __bf16* Qb   = (__bf16*)(ws + 60 * MB);  // 16 MB  [NH, S, HD]
    __bf16* Kb   = (__bf16*)(ws + 76 * MB);  //  4 MB  [NKV, S, HD]
    __bf16* Vtb  = (__bf16*)(ws + 80 * MB);  //  4 MB  [NKV, HD, S]
    __bf16* AttnB = Qraw;                    // Qraw dead after RoPE

    f32_to_bf16_kernel<<<2048, 256, 0, stream>>>(hs, Xb, (long)S_LEN * HID);
    f32_to_bf16_kernel<<<2048, 256, 0, stream>>>(Wq, Wqb, (long)NH * HD * HID);
    f32_to_bf16_kernel<<<1024, 256, 0, stream>>>(Wk, Wkb, (long)NKV * HD * HID);
    f32_to_bf16_kernel<<<1024, 256, 0, stream>>>(Wv, Wvb, (long)NKV * HD * HID);
    f32_to_bf16_kernel<<<2048, 256, 0, stream>>>(Wo, Wob, (long)HID * NH * HD);

    gemm_xwT<__bf16><<<dim3((NH * HD) / 128, S_LEN / 128), 256, 0, stream>>>(
        Xb, Wqb, Qraw, NH * HD, HID);
    gemm_xwT<__bf16><<<dim3((NKV * HD) / 128, S_LEN / 128), 256, 0, stream>>>(
        Xb, Wkb, Kraw, NKV * HD, HID);
    gemm_xwT<__bf16><<<dim3((NKV * HD) / 128, S_LEN / 128), 256, 0, stream>>>(
        Xb, Wvb, Vraw, NKV * HD, HID);

    rope_kernel<<<dim3(S_LEN, NH), 64, 0, stream>>>(Qraw, Qb, pos, NH);
    rope_kernel<<<dim3(S_LEN, NKV), 64, 0, stream>>>(Kraw, Kb, pos, NKV);
    vtrans_kernel<<<(S_LEN * NKV * HD + 255) / 256, 256, 0, stream>>>(Vraw, Vtb);

    attn_kernel<<<dim3(NH, S_LEN / 16), 32, 0, stream>>>(Qb, Kb, Vtb, AttnB);

    gemm_xwT<float><<<dim3(HID / 128, S_LEN / 128), 256, 0, stream>>>(
        AttnB, Wob, out, HID, NH * HD);
}
